// InfluenceDrivenController_34454227648746
// MI455X (gfx1250) — matrix operand order repaired and checked by reading, not compiled, not run on hardware
//
#include <hip/hip_runtime.h>
#include <hip/hip_bf16.h>

typedef __attribute__((ext_vector_type(16))) _Float16 v16h;
typedef __attribute__((ext_vector_type(8)))  float    v8f;

#define NROWS  100000
#define TOPK   1024
#define TIECAP 2048
#define SMEM_MAIN 188416

// ---------------- WMMA fragment helpers (CDNA5 16x16x32 f16 layouts) ----------------

__device__ __forceinline__ v16h frag_a(const _Float16* A, int lda, int row0, int k0, int lane) {
  // 16-bit A 16x32: lanes 0-15 row M=lane hold K 0-7 & 16-23; lanes 16-31 hold K 8-15 & 24-31
  int row = row0 + (lane & 15);
  int ko  = (lane >> 4) << 3;
  const _Float16* p = A + row * lda + k0 + ko;
  v16h a;
#pragma unroll
  for (int i = 0; i < 8; ++i) { a[i] = p[i]; a[i + 8] = p[i + 16]; }
  return a;
}

// B weights are pre-packed in fragment order: fragment f = (nt*KT + kt), lane L owns
// 16 contiguous halves at Wp[(f*32 + L)*16] == W[k0 + (L>>4)*8 + (e&7) + (e>>3)*16][n0 + (L&15)].
// One 32-byte per-lane load -> wave reads one contiguous 1KB block per K-step.

// O(f16)[16*Mtiles x Nout] = act(A(lds f16) * Wpacked + bias)
__device__ __forceinline__ void wmma_mlp_layer(
    const _Float16* A, int lda, const _Float16* Wp, const float* bias,
    _Float16* O, int ldo, int K, int Nout, int Mtiles,
    bool do_relu, int wid, int nwaves, int lane)
{
  int KT = K >> 5;
  int Ntiles = Nout >> 4;
  int total  = Mtiles * Ntiles;
  for (int t = wid; t < total; t += nwaves) {
    int mt = t / Ntiles, nt = t - mt * Ntiles;
    const v16h* bp = (const v16h*)Wp + ((size_t)nt * KT * 32 + lane);
    v8f acc = {};
    for (int kt = 0; kt < KT; ++kt) {
      v16h a = frag_a(A, lda, mt << 4, kt << 5, lane);
      v16h b = bp[(size_t)kt * 32];
      acc = __builtin_amdgcn_wmma_f32_16x16x32_f16(false, a, false, b, (short)0, acc, false, false);
    }
    int col   = (nt << 4) + (lane & 15);
    float bv  = bias[col];
    int mbase = (mt << 4) + ((lane >> 4) << 3);  // C tile: M = v + 8*lane[4], N = lane[3:0]
#pragma unroll
    for (int v = 0; v < 8; ++v) {
      float x = acc[v] + bv;
      if (do_relu) x = fmaxf(x, 0.f);
      O[(mbase + v) * ldo + col] = (_Float16)x;
    }
  }
}

// f32 output variant (pre-LayerNorm)
__device__ __forceinline__ void wmma_layer_f32(
    const _Float16* A, int lda, const _Float16* Wp, const float* bias,
    float* O, int ldo, int K, int Nout, int Mtiles,
    int wid, int nwaves, int lane)
{
  int KT = K >> 5;
  int Ntiles = Nout >> 4;
  int total  = Mtiles * Ntiles;
  for (int t = wid; t < total; t += nwaves) {
    int mt = t / Ntiles, nt = t - mt * Ntiles;
    const v16h* bp = (const v16h*)Wp + ((size_t)nt * KT * 32 + lane);
    v8f acc = {};
    for (int kt = 0; kt < KT; ++kt) {
      v16h a = frag_a(A, lda, mt << 4, kt << 5, lane);
      v16h b = bp[(size_t)kt * 32];
      acc = __builtin_amdgcn_wmma_f32_16x16x32_f16(false, a, false, b, (short)0, acc, false, false);
    }
    int col   = (nt << 4) + (lane & 15);
    float bv  = bias[col];
    int mbase = (mt << 4) + ((lane >> 4) << 3);
#pragma unroll
    for (int v = 0; v < 8; ++v) O[(mbase + v) * ldo + col] = acc[v] + bv;
  }
}

// ---------------- small prep kernels ----------------

__global__ void zero_u32_k(unsigned int* p, int n) {
  int i = blockIdx.x * blockDim.x + threadIdx.x;
  if (i < n) p[i] = 0u;
}

// Convert f32 weight [rows x cols] (row-major, K-major) to f16 packed fragment order,
// zero-padding K up to Kpad. dst has Kpad*cols halves.
__global__ void pack_b_k(_Float16* dst, const float* src, int rows, int cols, int Kpad) {
  int KT = Kpad >> 5;
  int total = (Kpad * cols);
  int i = blockIdx.x * blockDim.x + threadIdx.x;
  if (i >= total) return;
  int e    = i & 15;
  int lane = (i >> 4) & 31;
  int tile = i >> 9;            // nt*KT + kt
  int kt   = tile % KT;
  int nt   = tile / KT;
  int col  = (nt << 4) + (lane & 15);
  int k    = (kt << 5) + ((lane >> 4) << 3) + (e & 7) + ((e >> 3) << 4);
  dst[i] = (k < rows) ? (_Float16)src[(size_t)k * cols + col] : (_Float16)0.f;
}

// global encoder: Linear(16,32)-ReLU-Linear(32,64)-LayerNorm(64) -> gf[64]
__global__ void global_encoder_k(const float* gm, const float* w1, const float* b1,
                                 const float* w2, const float* b2,
                                 const float* g, const float* be, float* gf) {
  __shared__ float h1s[32];
  __shared__ float h2s[64];
  int t = threadIdx.x;  // 64 threads
  if (t < 32) {
    float s = b1[t];
    for (int k = 0; k < 16; ++k) s += gm[k] * w1[k * 32 + t];
    h1s[t] = fmaxf(s, 0.f);
  }
  __syncthreads();
  float y = b2[t];
  for (int k = 0; k < 32; ++k) y += h1s[k] * w2[k * 64 + t];
  h2s[t] = y;
  __syncthreads();
  float m = 0.f;
  for (int k = 0; k < 64; ++k) m += h2s[k];
  m *= (1.f / 64.f);
  float v = 0.f;
  for (int k = 0; k < 64; ++k) { float d = h2s[k] - m; v += d * d; }
  v *= (1.f / 64.f);
  gf[t] = (h2s[t] - m) * rsqrtf(v + 1e-5f) * g[t] + be[t];
}

// ---------------- fused main kernel: 64 rows per block, 8 waves ----------------

__global__ __launch_bounds__(256) void fused_main_k(
    const float* gnn, const float* world, const float* gf, const int* icv,
    const _Float16* fu1h, const float* fu1b,
    const _Float16* fu2h, const float* fu2b,
    const float* fu_g, const float* fu_be,
    const _Float16* sc1h, const float* sc1b,
    const _Float16* sc2h, const float* sc2b,
    const float* sc3w, const float* sc3b,
    const _Float16* va1h, const float* va1b,
    const _Float16* va2h, const float* va2b,
    const float* va3w, const float* va3b,
    _Float16* h2g, float* keys, float* influence, float* values)
{
  extern __shared__ char smem[];
  _Float16* Ah  = (_Float16*)smem;                 // [64][608] fused_in (f16)
  _Float16* H1  = (_Float16*)(smem + 77824);       // [64][384]
  float*    H2f = (float*)(smem + 126976);         // [64][128] pre-LN
  _Float16* H2h = (_Float16*)(smem + 159744);      // [64][128] post-LN
  _Float16* S1  = (_Float16*)(smem + 176128);      // [64][64]
  _Float16* S2  = (_Float16*)(smem + 184320);      // [64][32]

  int tid  = threadIdx.x;
  int wid  = tid >> 5, lane = tid & 31;
  int row0 = blockIdx.x * 64;

  // ---- stage fused_in = [gnn(256) | gf(64) | mean(world,5)(257) | pad(31)] as f16 ----
  for (int i = tid; i < 64 * 256; i += 256) {
    int r = i >> 8, c = i & 255, gr = row0 + r;
    Ah[r * 608 + c] = (gr < NROWS) ? (_Float16)gnn[(long)gr * 256 + c] : (_Float16)0.f;
  }
  for (int i = tid; i < 64 * 64; i += 256) {
    int r = i >> 6, c = i & 63, gr = row0 + r;
    Ah[r * 608 + 256 + c] = (gr < NROWS) ? (_Float16)gf[c] : (_Float16)0.f;
  }
  for (int i = tid; i < 64 * 257; i += 256) {
    int r = i / 257, c = i - r * 257, gr = row0 + r;
    float s = 0.f;
    if (gr < NROWS) {
      long base = (long)gr * 1285 + c;  // 5*257
      s = world[base] + world[base + 257] + world[base + 514] + world[base + 771] + world[base + 1028];
      s *= 0.2f;
    }
    Ah[r * 608 + 320 + c] = (_Float16)s;
  }
  for (int i = tid; i < 64 * 31; i += 256) {
    int r = i / 31, c = i - r * 31;
    Ah[r * 608 + 577 + c] = (_Float16)0.f;
  }
  __syncthreads();

  // ---- fu1: [64x608]x[608x384] relu ; fu2: [64x384]x[384x128] + bias ----
  wmma_mlp_layer(Ah, 608, fu1h, fu1b, H1, 384, 608, 384, 4, true, wid, 8, lane);
  __syncthreads();
  wmma_layer_f32(H1, 384, fu2h, fu2b, H2f, 128, 384, 128, 4, wid, 8, lane);
  __syncthreads();

  // ---- LayerNorm(128) per row; keep f16 in LDS + store to global for gather ----
  if (tid < 64) {
    int r = tid, gr = row0 + r;
    float m = 0.f;
    for (int c = 0; c < 128; ++c) m += H2f[r * 128 + c];
    m *= (1.f / 128.f);
    float v = 0.f;
    for (int c = 0; c < 128; ++c) { float d = H2f[r * 128 + c] - m; v += d * d; }
    v *= (1.f / 128.f);
    float rs = rsqrtf(v + 1e-5f);
    for (int c = 0; c < 128; ++c) {
      float y = (H2f[r * 128 + c] - m) * rs * fu_g[c] + fu_be[c];
      H2h[r * 128 + c] = (_Float16)y;
      if (gr < NROWS) h2g[(long)gr * 128 + c] = (_Float16)y;
    }
  }
  __syncthreads();

  // ---- score head: 128->64->32->1 (sigmoid) ----
  wmma_mlp_layer(H2h, 128, sc1h, sc1b, S1, 64, 128, 64, 4, true, wid, 8, lane);
  __syncthreads();
  wmma_mlp_layer(S1, 64, sc2h, sc2b, S2, 32, 64, 32, 4, true, wid, 8, lane);
  __syncthreads();
  float score = 0.f;
  if (tid < 64) {
    float s = sc3b[0];
    for (int c = 0; c < 32; ++c) s += (float)S2[tid * 32 + c] * sc3w[c];
    score = 1.f / (1.f + expf(-s));
  }
  __syncthreads();

  // ---- value head: 128->64->32->1 (identity), reuse S1/S2 ----
  wmma_mlp_layer(H2h, 128, va1h, va1b, S1, 64, 128, 64, 4, true, wid, 8, lane);
  __syncthreads();
  wmma_mlp_layer(S1, 64, va2h, va2b, S2, 32, 64, 32, 4, true, wid, 8, lane);
  __syncthreads();
  if (tid < 64) {
    int gr = row0 + tid;
    if (gr < NROWS) {
      float v = va3b[0];
      for (int c = 0; c < 32; ++c) v += (float)S2[tid * 32 + c] * va3w[c];
      values[gr] = v;
      bool ic = icv[gr] != 0;
      influence[gr] = ic ? score : 0.f;
      keys[gr] = ic ? score : -__builtin_inff();
    }
  }
}

// ---------------- top-k: two-level radix histogram on positive-float bit order ----------------

__global__ void hist1_k(const float* keys, unsigned int* hist1) {
  for (int i = blockIdx.x * blockDim.x + threadIdx.x; i < NROWS; i += gridDim.x * blockDim.x) {
    float k = keys[i];
    if (k >= 0.f) atomicAdd(&hist1[__float_as_uint(k) >> 20], 1u);  // sigmoid in (0,1) -> bucket < 1016
  }
}

__global__ void scan1_k(const unsigned int* hist1, int* scal) {
  if (threadIdx.x == 0 && blockIdx.x == 0) {
    int cum = 0, T1 = 0, above = 0; bool found = false;
    for (int b = 1023; b >= 0; --b) {
      int h = (int)hist1[b];
      if (cum + h >= TOPK) { T1 = b; above = cum; found = true; break; }
      cum += h;
    }
    if (!found) { T1 = 0; above = cum - (int)hist1[0]; }
    scal[0] = T1; scal[1] = above;
  }
}

__global__ void hist2_k(const float* keys, const int* scal, unsigned int* hist2) {
  int T1 = scal[0];
  for (int i = blockIdx.x * blockDim.x + threadIdx.x; i < NROWS; i += gridDim.x * blockDim.x) {
    float k = keys[i];
    if (k < 0.f) continue;
    unsigned bits = __float_as_uint(k);
    if ((int)(bits >> 20) == T1) atomicAdd(&hist2[(bits >> 9) & 2047], 1u);
  }
}

__global__ void scan2_k(const unsigned int* hist2, int* scal) {
  if (threadIdx.x == 0 && blockIdx.x == 0) {
    int target = TOPK - scal[1];
    if (target <= 0) { scal[2] = 2048; scal[3] = scal[1]; return; }
    int cum = 0, T2 = 0, above = 0; bool found = false;
    for (int b = 2047; b >= 0; --b) {
      int h = (int)hist2[b];
      if (cum + h >= target) { T2 = b; above = cum; found = true; break; }
      cum += h;
    }
    if (!found) { T2 = 0; above = cum - (int)hist2[0]; }
    scal[2] = T2; scal[3] = scal[1] + above;
  }
}

__global__ void collect_k(const float* keys, const int* scal, unsigned int* cnt,
                          float* selS, int* selI, float* tieS, int* tieI) {
  int T1 = scal[0], T2 = scal[2];
  for (int i = blockIdx.x * blockDim.x + threadIdx.x; i < NROWS; i += gridDim.x * blockDim.x) {
    float k = keys[i];
    if (!(k >= 0.f)) continue;
    unsigned bits = __float_as_uint(k);
    int b1 = (int)(bits >> 20);
    if (b1 > T1) {
      unsigned p = atomicAdd(&cnt[0], 1u);
      if (p < TOPK) { selS[p] = k; selI[p] = i; }
    } else if (b1 == T1) {
      int b2 = (int)((bits >> 9) & 2047);
      if (b2 > T2) {
        unsigned p = atomicAdd(&cnt[0], 1u);
        if (p < TOPK) { selS[p] = k; selI[p] = i; }
      } else if (b2 == T2) {
        unsigned p = atomicAdd(&cnt[1], 1u);
        if (p < TIECAP) { tieS[p] = k; tieI[p] = i; }
      }
    }
  }
}

__device__ __forceinline__ bool later_pair(float sa, int ia, float sb, int ib) {
  // "a sorts after b" under (score desc, idx asc)
  return (sa < sb) || (sa == sb && ia > ib);
}

__global__ __launch_bounds__(1024) void finalize_k(
    const unsigned int* cnt, const int* scal,
    const float* selS, const int* selI, const float* tieS, const int* tieI,
    float* topScores, int* topIdx)
{
  __shared__ float ks[TOPK];   __shared__ int is[TOPK];
  __shared__ float ts[TIECAP]; __shared__ int ti[TIECAP];
  int tid = threadIdx.x;
  int W0 = (int)cnt[0]; if (W0 > TOPK) W0 = TOPK;
  int M  = (int)cnt[1]; if (M  > TIECAP) M = TIECAP;
  (void)scal;

  for (int i = tid; i < TIECAP; i += 1024) {
    if (i < M) { ts[i] = tieS[i]; ti[i] = tieI[i]; }
    else       { ts[i] = -__builtin_inff(); ti[i] = 0x7fffffff; }
  }
  __syncthreads();
  // bitonic sort ties by (score desc, idx asc)
  for (int k = 2; k <= TIECAP; k <<= 1)
    for (int j = k >> 1; j > 0; j >>= 1) {
      for (int i = tid; i < TIECAP; i += 1024) {
        int ixj = i ^ j;
        if (ixj > i) {
          bool up = ((i & k) == 0);
          bool lat = later_pair(ts[i], ti[i], ts[ixj], ti[ixj]);
          if (up ? lat : !lat) {
            float f = ts[i]; ts[i] = ts[ixj]; ts[ixj] = f;
            int   t = ti[i]; ti[i] = ti[ixj]; ti[ixj] = t;
          }
        }
      }
      __syncthreads();
    }
  int need = TOPK - W0; if (need > M) need = M; if (need < 0) need = 0;
  for (int i = tid; i < TOPK; i += 1024) {
    if (i < W0)            { ks[i] = selS[i];      is[i] = selI[i]; }
    else if (i < W0 + need){ ks[i] = ts[i - W0];   is[i] = ti[i - W0]; }
    else                   { ks[i] = -__builtin_inff(); is[i] = 0x7fffffff; }
  }
  __syncthreads();
  // bitonic sort final 1024 by (score desc, idx asc)
  for (int k = 2; k <= TOPK; k <<= 1)
    for (int j = k >> 1; j > 0; j >>= 1) {
      int i = tid, ixj = i ^ j;
      if (ixj > i) {
        bool up = ((i & k) == 0);
        bool lat = later_pair(ks[i], is[i], ks[ixj], is[ixj]);
        if (up ? lat : !lat) {
          float f = ks[i]; ks[i] = ks[ixj]; ks[ixj] = f;
          int   t = is[i]; is[i] = is[ixj]; is[ixj] = t;
        }
      }
      __syncthreads();
    }
  for (int i = tid; i < TOPK; i += 1024) { topScores[i] = ks[i]; topIdx[i] = is[i]; }
}

// ---------------- action heads on the top-K rows ----------------

__global__ __launch_bounds__(128) void actions_k(
    const _Float16* h2g, const int* topIdx,
    const _Float16* ac1h, const float* ac1b, const _Float16* ac2h, const float* ac2b,
    const float* ac3w, const float* ac3b,
    const _Float16* lc1h, const float* lc1b, const _Float16* lc2h, const float* lc2b,
    const float* lc3w, const float* lc3b, float* raw)
{
  __shared__ _Float16 Ah[16 * 128];
  __shared__ _Float16 S1a[16 * 64], S2a[16 * 32], S1l[16 * 64], S2l[16 * 32];
  int tid = threadIdx.x, wid = tid >> 5, lane = tid & 31;
  int row0 = blockIdx.x * 16;
  for (int i = tid; i < 16 * 128; i += 128) {
    int r = i >> 7, c = i & 127;
    int gi = topIdx[row0 + r];
    Ah[i] = (gi >= 0 && gi < NROWS) ? h2g[(long)gi * 128 + c] : (_Float16)0.f;
  }
  __syncthreads();
  wmma_mlp_layer(Ah, 128, ac1h, ac1b, S1a, 64, 128, 64, 1, true, wid, 4, lane);
  wmma_mlp_layer(Ah, 128, lc1h, lc1b, S1l, 64, 128, 64, 1, true, wid, 4, lane);
  __syncthreads();
  wmma_mlp_layer(S1a, 64, ac2h, ac2b, S2a, 32, 64, 32, 1, true, wid, 4, lane);
  wmma_mlp_layer(S1l, 64, lc2h, lc2b, S2l, 32, 64, 32, 1, true, wid, 4, lane);
  __syncthreads();
  if (tid < 16) {
    float a = ac3b[0], l = lc3b[0];
    for (int c = 0; c < 32; ++c) {
      a += (float)S2a[tid * 32 + c] * ac3w[c];
      l += (float)S2l[tid * 32 + c] * lc3w[c];
    }
    a = tanhf(a);
    l = 1.f / (1.f + expf(-l));
    raw[(row0 + tid) * 2 + 0] = a;
    raw[(row0 + tid) * 2 + 1] = l;
  }
}

// ---------------- host launcher ----------------

static inline size_t alignup(size_t x, size_t a) { return (x + a - 1) & ~(a - 1); }

extern "C" void kernel_launch(void* const* d_in, const int* in_sizes, int n_in,
                              void* d_out, int out_size, void* d_ws, size_t ws_size,
                              hipStream_t stream) {
  (void)in_sizes; (void)n_in; (void)out_size; (void)ws_size;
  const float* gnn   = (const float*)d_in[0];
  const float* world = (const float*)d_in[1];
  const float* gm    = (const float*)d_in[2];
  const int*   icv   = (const int*)d_in[3];
  const float *ge1w = (const float*)d_in[4],  *ge1b = (const float*)d_in[5];
  const float *ge2w = (const float*)d_in[6],  *ge2b = (const float*)d_in[7];
  const float *fu1w = (const float*)d_in[8],  *fu1b = (const float*)d_in[9];
  const float *fu2w = (const float*)d_in[10], *fu2b = (const float*)d_in[11];
  const float *sc1w = (const float*)d_in[12], *sc1b = (const float*)d_in[13];
  const float *sc2w = (const float*)d_in[14], *sc2b = (const float*)d_in[15];
  const float *sc3w = (const float*)d_in[16], *sc3b = (const float*)d_in[17];
  const float *ac1w = (const float*)d_in[18], *ac1b = (const float*)d_in[19];
  const float *ac2w = (const float*)d_in[20], *ac2b = (const float*)d_in[21];
  const float *ac3w = (const float*)d_in[22], *ac3b = (const float*)d_in[23];
  const float *lc1w = (const float*)d_in[24], *lc1b = (const float*)d_in[25];
  const float *lc2w = (const float*)d_in[26], *lc2b = (const float*)d_in[27];
  const float *lc3w = (const float*)d_in[28], *lc3b = (const float*)d_in[29];
  const float *va1w = (const float*)d_in[30], *va1b = (const float*)d_in[31];
  const float *va2w = (const float*)d_in[32], *va2b = (const float*)d_in[33];
  const float *va3w = (const float*)d_in[34], *va3b = (const float*)d_in[35];
  const float *ge_g = (const float*)d_in[36], *ge_be = (const float*)d_in[37];
  const float *fu_g = (const float*)d_in[38], *fu_be = (const float*)d_in[39];

  char* ws = (char*)d_ws;
  size_t cur = 0;
  auto take = [&](size_t bytes) -> char* {
    char* p = ws + cur;
    cur = alignup(cur + bytes, 256);
    return p;
  };
  float*    gfb  = (float*)take(64 * 4);
  _Float16* fu1h = (_Float16*)take((size_t)608 * 384 * 2);
  _Float16* fu2h = (_Float16*)take((size_t)384 * 128 * 2);
  _Float16* sc1h = (_Float16*)take(128 * 64 * 2);
  _Float16* sc2h = (_Float16*)take(64 * 32 * 2);
  _Float16* ac1h = (_Float16*)take(128 * 64 * 2);
  _Float16* ac2h = (_Float16*)take(64 * 32 * 2);
  _Float16* lc1h = (_Float16*)take(128 * 64 * 2);
  _Float16* lc2h = (_Float16*)take(64 * 32 * 2);
  _Float16* va1h = (_Float16*)take(128 * 64 * 2);
  _Float16* va2h = (_Float16*)take(64 * 32 * 2);
  float*    keys = (float*)take((size_t)NROWS * 4);
  unsigned int* hist1 = (unsigned int*)take((1024 + 2048 + 8) * 4);  // hist1|hist2|cnt contiguous
  unsigned int* hist2 = hist1 + 1024;
  unsigned int* cnt   = hist2 + 2048;
  int*      scal = (int*)take(32);
  float*    selS = (float*)take(TOPK * 4);
  int*      selI = (int*)take(TOPK * 4);
  float*    tieS = (float*)take(TIECAP * 4);
  int*      tieI = (int*)take(TIECAP * 4);
  _Float16* h2g  = (_Float16*)take((size_t)NROWS * 128 * 2);

  float* out  = (float*)d_out;
  float* rawA = out;                                // [1024,2]
  float* infl = out + 2048;                         // [N]
  float* vals = out + 2048 + NROWS;                 // [N]
  float* topS = out + 2048 + 2 * NROWS;             // [1024]
  int*   topI = (int*)(out + 2048 + 2 * NROWS + TOPK);  // [1024] int bits

  zero_u32_k<<<(1024 + 2048 + 8 + 255) / 256, 256, 0, stream>>>(hist1, 1024 + 2048 + 8);
  global_encoder_k<<<1, 64, 0, stream>>>(gm, ge1w, ge1b, ge2w, ge2b, ge_g, ge_be, gfb);

  // pack all GEMM weights into WMMA fragment order (f16), padding K where needed
  pack_b_k<<<(608 * 384 + 255) / 256, 256, 0, stream>>>(fu1h, fu1w, 577, 384, 608);
  pack_b_k<<<(384 * 128 + 255) / 256, 256, 0, stream>>>(fu2h, fu2w, 384, 128, 384);
  pack_b_k<<<32, 256, 0, stream>>>(sc1h, sc1w, 128, 64, 128);
  pack_b_k<<<8,  256, 0, stream>>>(sc2h, sc2w, 64, 32, 64);
  pack_b_k<<<32, 256, 0, stream>>>(ac1h, ac1w, 128, 64, 128);
  pack_b_k<<<8,  256, 0, stream>>>(ac2h, ac2w, 64, 32, 64);
  pack_b_k<<<32, 256, 0, stream>>>(lc1h, lc1w, 128, 64, 128);
  pack_b_k<<<8,  256, 0, stream>>>(lc2h, lc2w, 64, 32, 64);
  pack_b_k<<<32, 256, 0, stream>>>(va1h, va1w, 128, 64, 128);
  pack_b_k<<<8,  256, 0, stream>>>(va2h, va2w, 64, 32, 64);

  fused_main_k<<<(NROWS + 63) / 64, 256, SMEM_MAIN, stream>>>(
      gnn, world, gfb, icv,
      fu1h, fu1b, fu2h, fu2b, fu_g, fu_be,
      sc1h, sc1b, sc2h, sc2b, sc3w, sc3b,
      va1h, va1b, va2h, va2b, va3w, va3b,
      h2g, keys, infl, vals);

  hist1_k<<<512, 256, 0, stream>>>(keys, hist1);
  scan1_k<<<1, 32, 0, stream>>>(hist1, scal);
  hist2_k<<<512, 256, 0, stream>>>(keys, scal, hist2);
  scan2_k<<<1, 32, 0, stream>>>(hist2, scal);
  collect_k<<<512, 256, 0, stream>>>(keys, scal, cnt, selS, selI, tieS, tieI);
  finalize_k<<<1, 1024, 0, stream>>>(cnt, scal, selS, selI, tieS, tieI, topS, topI);

  actions_k<<<TOPK / 16, 128, 0, stream>>>(h2g, topI,
      ac1h, ac1b, ac2h, ac2b, ac3w, ac3b,
      lc1h, lc1b, lc2h, lc2b, lc3w, lc3b, rawA);
}
